// GATNet_17102559772862
// MI455X (gfx1250) — matrix operand order repaired and checked by reading, not compiled, not run on hardware
//
#include <hip/hip_runtime.h>

// ---------------------------------------------------------------------------
// GATNet forward on MI455X (gfx1250, wave32).
// GEMMs: bf16 WMMA (v_wmma_f32_16x16x32_bf16), f32 accumulate.
//   - block = 8 waves = 128 rows x 64 cols; packed B panel staged once per
//     block into LDS with global_load_async_to_lds_b128 (ASYNCcnt path),
//     then read back with ds_load_b128. Cuts B-side L2 traffic 8x.
// Edge phase: f32 scalar + global atomics (memory/atomic bound anyway).
// ---------------------------------------------------------------------------

typedef __attribute__((ext_vector_type(16))) __bf16     v16bf;
typedef __attribute__((ext_vector_type(8)))  float      v8f;
typedef __attribute__((ext_vector_type(4)))  unsigned   u32x4;

union Frag16 { v16bf v; u32x4 q[2]; };

__device__ __forceinline__ unsigned short bf16_of_f32(float f) {
    unsigned int u = __float_as_uint(f);
    u += 0x7FFFu + ((u >> 16) & 1u);      // round-to-nearest-even
    return (unsigned short)(u >> 16);
}

// ---------------------------------------------------------------------------
__global__ void cast_bf16_kernel(const float* __restrict__ in,
                                 unsigned short* __restrict__ out, int n) {
    int i = blockIdx.x * blockDim.x + threadIdx.x;
    if (i < n) out[i] = bf16_of_f32(in[i]);
}

__global__ void bias_cast_bf16_kernel(const float* __restrict__ x,
                                      const float* __restrict__ b,
                                      unsigned short* __restrict__ out,
                                      int n, int D) {
    int i = blockIdx.x * blockDim.x + threadIdx.x;
    if (i < n) out[i] = bf16_of_f32(x[i] + b[i % D]);
}

__global__ void fill_f32_kernel(float* __restrict__ p, float v, int n) {
    int i = blockIdx.x * blockDim.x + threadIdx.x;
    if (i < n) p[i] = v;
}

// ---------------------------------------------------------------------------
// Repack a row-major f32 weight [K, Nc] into the WMMA 16-bit B-operand layout:
// per 32x16 k-tile, 512 contiguous u16:  idx = (kt*(Nc/16)+nt)*512 + lane*16 + j
// lane<16 holds K rows (kt*32+0..7, +16..23), lane>=16 holds rows (kt*32+8..15, +24..31).
__global__ void repack_weight_bf16_kernel(const float* __restrict__ W,
                                          unsigned short* __restrict__ Bp,
                                          int K, int Nc) {
    int i = blockIdx.x * blockDim.x + threadIdx.x;
    if (i >= K * Nc) return;
    int j    = i & 15;
    int lane = (i >> 4) & 31;
    int tile = i >> 9;
    int tn   = tile % (Nc >> 4);
    int kt   = tile / (Nc >> 4);
    int krow = (kt << 5) + ((lane >> 4) << 3) + (j & 7) + ((j >> 3) << 4);
    int col  = (tn << 4) + (lane & 15);
    Bp[i] = bf16_of_f32(W[(size_t)krow * Nc + col]);
}

// ---------------------------------------------------------------------------
// C[M,Nc] = A[M,K] @ B[K,Nc]; A row-major bf16, Bp pre-packed bf16, C f32.
// grid = (Nc/64, ceil(Mtiles/8)); block = 256 threads = 8 waves.
// All waves of a block share one 64-column B panel staged in LDS.
template <int K>
__global__ __launch_bounds__(256)
void gemm_bf16_wmma(const unsigned short* __restrict__ A,
                    const unsigned short* __restrict__ Bp,
                    float* __restrict__ C, int M, int Nc) {
    constexpr int KT = K / 32;                       // k-chunks
    __shared__ __align__(32) unsigned short ldsB[KT * 2048];  // KT * 4KB

    const int lane = threadIdx.x & 31;
    const int wave = threadIdx.x >> 5;
    const int tn4  = blockIdx.x;                     // 64-wide N macro tile
    const int tm   = blockIdx.y * 8 + wave;          // 16-row M tile per wave
    const bool active = (tm < (M >> 4));             // wave-uniform

    // ---- stage packed B panel into LDS (async DMA, ASYNCcnt) -------------
    {
        const unsigned lds_base = (unsigned)(size_t)&ldsB[0];
        const unsigned lds_off  = lds_base + threadIdx.x * 16u;
        const unsigned short* gsrc =
            Bp + (size_t)(tn4 * 4) * 512 + threadIdx.x * 8;  // +16 bytes/lane
        const size_t gstride = (size_t)(Nc >> 4) * 512;      // u16 per k-chunk
#pragma unroll
        for (int kt = 0; kt < KT; ++kt) {
            unsigned long long ga = (unsigned long long)(gsrc + kt * gstride);
            unsigned la = lds_off + (unsigned)kt * 4096u;
            asm volatile("global_load_async_to_lds_b128 %0, %1, off"
                         :: "v"(la), "v"(ga) : "memory");
        }
        asm volatile("s_wait_asynccnt 0x0" ::: "memory");
    }
    __syncthreads();

    if (active) {
        // A fragment (16-bit A layout): lane<16 -> K = 0..7,16..23 per chunk
        const int row  = (tm << 4) + (lane & 15);
        const int koff = (lane >> 4) << 3;           // 0 or 8
        const unsigned short* Ar = A + (size_t)row * K + koff;
        const unsigned short* lb = &ldsB[lane * 16];

        v8f acc0 = {}, acc1 = {}, acc2 = {}, acc3 = {};

#pragma unroll 4
        for (int kt = 0; kt < KT; ++kt) {
            Frag16 a;
            a.q[0] = *(const u32x4*)(Ar + kt * 32);       // K = koff   ..+7
            a.q[1] = *(const u32x4*)(Ar + kt * 32 + 16);  // K = koff+16..+23
            const unsigned short* lk = lb + kt * 2048;
            Frag16 b0, b1, b2, b3;
            b0.q[0] = *(const u32x4*)(lk + 0 * 512);
            b0.q[1] = *(const u32x4*)(lk + 0 * 512 + 8);
            b1.q[0] = *(const u32x4*)(lk + 1 * 512);
            b1.q[1] = *(const u32x4*)(lk + 1 * 512 + 8);
            b2.q[0] = *(const u32x4*)(lk + 2 * 512);
            b2.q[1] = *(const u32x4*)(lk + 2 * 512 + 8);
            b3.q[0] = *(const u32x4*)(lk + 3 * 512);
            b3.q[1] = *(const u32x4*)(lk + 3 * 512 + 8);
            acc0 = __builtin_amdgcn_wmma_f32_16x16x32_bf16(false, a.v, false, b0.v,
                                                           (short)0, acc0, false, false);
            acc1 = __builtin_amdgcn_wmma_f32_16x16x32_bf16(false, a.v, false, b1.v,
                                                           (short)0, acc1, false, false);
            acc2 = __builtin_amdgcn_wmma_f32_16x16x32_bf16(false, a.v, false, b2.v,
                                                           (short)0, acc2, false, false);
            acc3 = __builtin_amdgcn_wmma_f32_16x16x32_bf16(false, a.v, false, b3.v,
                                                           (short)0, acc3, false, false);
        }

        // C/D layout: VGPR r, lane L -> M = r + 8*(L/16), N = L%16
        const int crow = (tm << 4) + ((lane >> 4) << 3);
        const int ccol = (tn4 << 6) + (lane & 15);
        float* Cp = C + (size_t)crow * Nc + ccol;
#pragma unroll
        for (int r = 0; r < 8; ++r) {
            Cp[(size_t)r * Nc + 0]  = acc0[r];
            Cp[(size_t)r * Nc + 16] = acc1[r];
            Cp[(size_t)r * Nc + 32] = acc2[r];
            Cp[(size_t)r * Nc + 48] = acc3[r];
        }
    }
}

// ---------------------------------------------------------------------------
// Edge list with PyG-style self loops appended
__global__ void build_edges_kernel(const long long* __restrict__ ei,
                                   int* __restrict__ src, int* __restrict__ dst,
                                   int E, int Etot) {
    int i = blockIdx.x * blockDim.x + threadIdx.x;
    if (i >= Etot) return;
    if (i < E) { src[i] = (int)ei[i]; dst[i] = (int)ei[(size_t)E + i]; }
    else       { src[i] = i - E;      dst[i] = i - E; }
}

// Per-(node,head) attention logits, H=8 C=64
__global__ void attn_logits_h8_kernel(const float* __restrict__ h,
                                      const float* __restrict__ a_s,
                                      const float* __restrict__ a_d,
                                      float* __restrict__ als,
                                      float* __restrict__ ald, int n) {
    int i = blockIdx.x * blockDim.x + threadIdx.x;
    if (i >= n) return;
    int node = i >> 3, head = i & 7;
    const float* hp  = h + (size_t)node * 512 + head * 64;
    const float* asp = a_s + head * 64;
    const float* adp = a_d + head * 64;
    float s = 0.f, d = 0.f;
#pragma unroll 8
    for (int c = 0; c < 64; ++c) { float v = hp[c]; s += v * asp[c]; d += v * adp[c]; }
    als[i] = s; ald[i] = d;
}

// Per-node attention logits, H=1 C=512
__global__ void attn_logits_h1_kernel(const float* __restrict__ h,
                                      const float* __restrict__ a_s,
                                      const float* __restrict__ a_d,
                                      float* __restrict__ als,
                                      float* __restrict__ ald, int N) {
    int i = blockIdx.x * blockDim.x + threadIdx.x;
    if (i >= N) return;
    const float* hp = h + (size_t)i * 512;
    float s = 0.f, d = 0.f;
    for (int c = 0; c < 512; ++c) { float v = hp[c]; s += v * a_s[c]; d += v * a_d[c]; }
    als[i] = s; ald[i] = d;
}

// e[e,h] = leaky_relu(als[src] + ald[dst])
__global__ void edge_score_kernel(const int* __restrict__ src,
                                  const int* __restrict__ dst,
                                  const float* __restrict__ als,
                                  const float* __restrict__ ald,
                                  float* __restrict__ e, int Etot, int H) {
    int i = blockIdx.x * blockDim.x + threadIdx.x;
    if (i >= Etot * H) return;
    int ei = i / H, h = i - ei * H;
    float v = als[(size_t)src[ei] * H + h] + ald[(size_t)dst[ei] * H + h];
    e[i] = (v > 0.f) ? v : 0.2f * v;
}

__device__ __forceinline__ void atomic_max_f32(float* addr, float val) {
    int* ai = (int*)addr;
    int old = __hip_atomic_load(ai, __ATOMIC_RELAXED, __HIP_MEMORY_SCOPE_AGENT);
    while (__int_as_float(old) < val) {
        int assumed = old;
        old = atomicCAS(ai, assumed, __float_as_int(val));
        if (old == assumed) break;
    }
}

__global__ void seg_max_kernel(const int* __restrict__ dst,
                               const float* __restrict__ e,
                               float* __restrict__ m, int Etot, int H) {
    int i = blockIdx.x * blockDim.x + threadIdx.x;
    if (i >= Etot * H) return;
    int ei = i / H, h = i - ei * H;
    atomic_max_f32(&m[(size_t)dst[ei] * H + h], e[i]);
}

__global__ void seg_expsum_kernel(const int* __restrict__ dst,
                                  float* __restrict__ e,
                                  const float* __restrict__ m,
                                  float* __restrict__ s, int Etot, int H) {
    int i = blockIdx.x * blockDim.x + threadIdx.x;
    if (i >= Etot * H) return;
    int ei = i / H, h = i - ei * H;
    float ex = __expf(e[i] - m[(size_t)dst[ei] * H + h]);
    e[i] = ex;
    atomicAdd(&s[(size_t)dst[ei] * H + h], ex);
}

// out[dst] += h[src] * alpha   (one 256-thread block per edge, D=512)
__global__ __launch_bounds__(256)
void aggregate_kernel(const int* __restrict__ src, const int* __restrict__ dst,
                      const float* __restrict__ h, const float* __restrict__ ex,
                      const float* __restrict__ s, float* __restrict__ out,
                      int Etot, int H, int C) {
    int e = blockIdx.x;
    if (e >= Etot) return;
    int sn = src[e], dn = dst[e];
    const int D = H * C;
    for (int d = threadIdx.x; d < D; d += blockDim.x) {
        int head = d / C;
        float alpha = ex[(size_t)e * H + head] / s[(size_t)dn * H + head];
        atomicAdd(&out[(size_t)dn * D + d], h[(size_t)sn * D + d] * alpha);
    }
}

// y = gamma*(relu(x+bias)-mu)*rsqrt(var+eps)+beta; one block per channel.
__global__ __launch_bounds__(256)
void bn_relu_kernel(const float* __restrict__ x, const float* __restrict__ bias,
                    const float* __restrict__ gamma, const float* __restrict__ beta,
                    float* __restrict__ yf, unsigned short* __restrict__ ybf,
                    int Nrows, int D) {
    const int c = blockIdx.x;
    __shared__ float s1[256], s2[256];
    float sum = 0.f, sq = 0.f;
    const float bc = bias[c];
    for (int r = threadIdx.x; r < Nrows; r += blockDim.x) {
        float v = x[(size_t)r * D + c] + bc;
        v = v > 0.f ? v : 0.f;
        sum += v; sq += v * v;
    }
    s1[threadIdx.x] = sum; s2[threadIdx.x] = sq;
    __syncthreads();
    for (int off = 128; off > 0; off >>= 1) {
        if (threadIdx.x < off) {
            s1[threadIdx.x] += s1[threadIdx.x + off];
            s2[threadIdx.x] += s2[threadIdx.x + off];
        }
        __syncthreads();
    }
    const float inv_n = 1.0f / (float)Nrows;
    const float mu  = s1[0] * inv_n;
    const float var = s2[0] * inv_n - mu * mu;     // biased, like torch BN train
    const float sc  = rsqrtf(var + 1e-5f) * gamma[c];
    const float sh  = beta[c] - mu * sc;
    for (int r = threadIdx.x; r < Nrows; r += blockDim.x) {
        float v = x[(size_t)r * D + c] + bc;
        v = v > 0.f ? v : 0.f;
        float y = v * sc + sh;
        yf[(size_t)r * D + c]  = y;
        ybf[(size_t)r * D + c] = bf16_of_f32(y);
    }
}

// out = relu(h @ lin_w + lin_b), lin_w [512,2]
__global__ void final_linear_kernel(const float* __restrict__ h,
                                    const float* __restrict__ w,
                                    const float* __restrict__ b,
                                    float* __restrict__ out, int N, int O) {
    int i = blockIdx.x * blockDim.x + threadIdx.x;
    if (i >= N * O) return;
    int n = i / O, o = i - n * O;
    const float* hp = h + (size_t)n * 512;
    float acc = b[o];
    for (int d = 0; d < 512; ++d) acc += hp[d] * w[d * O + o];
    out[i] = acc > 0.f ? acc : 0.f;
}

// ---------------------------------------------------------------------------
extern "C" void kernel_launch(void* const* d_in, const int* in_sizes, int n_in,
                              void* d_out, int out_size, void* d_ws, size_t ws_size,
                              hipStream_t stream) {
    const float*     x      = (const float*)d_in[0];
    const long long* eidx   = (const long long*)d_in[1];
    const float*     W1     = (const float*)d_in[2];
    const float*     a1s    = (const float*)d_in[3];
    const float*     a1d    = (const float*)d_in[4];
    const float*     b1     = (const float*)d_in[5];
    const float*     W2     = (const float*)d_in[6];
    const float*     a2s    = (const float*)d_in[7];
    const float*     a2d    = (const float*)d_in[8];
    const float*     b2     = (const float*)d_in[9];
    const float*     fc1_w  = (const float*)d_in[10];
    const float*     fc1_b  = (const float*)d_in[11];
    const float*     g1     = (const float*)d_in[12];
    const float*     be1    = (const float*)d_in[13];
    const float*     fc2_w  = (const float*)d_in[14];
    const float*     fc2_b  = (const float*)d_in[15];
    const float*     g2     = (const float*)d_in[16];
    const float*     be2    = (const float*)d_in[17];
    const float*     lin_w  = (const float*)d_in[18];
    const float*     lin_b  = (const float*)d_in[19];
    float*           out    = (float*)d_out;

    const int Fin  = 128, D = 512, OUT = 2;
    const int N    = in_sizes[0] / Fin;
    const int E    = in_sizes[1] / 2;
    const int Etot = E + N;

    // ---- carve workspace -------------------------------------------------
    char* wp = (char*)d_ws;
    auto alloc = [&](size_t bytes) -> char* {
        char* r = wp; wp += (bytes + 255) & ~(size_t)255; return r;
    };
    int*            srcI  = (int*)alloc(sizeof(int) * (size_t)Etot);
    int*            dstI  = (int*)alloc(sizeof(int) * (size_t)Etot);
    unsigned short* xb    = (unsigned short*)alloc(2ull * N * Fin);
    unsigned short* w1p   = (unsigned short*)alloc(2ull * Fin * D);
    unsigned short* fc1p  = (unsigned short*)alloc(2ull * D * D);
    unsigned short* w2p   = (unsigned short*)alloc(2ull * D * D);
    unsigned short* fc2p  = (unsigned short*)alloc(2ull * D * D);
    float*          bufA  = (float*)alloc(4ull * N * D);
    float*          bufB  = (float*)alloc(4ull * N * D);
    float*          bufC  = (float*)alloc(4ull * N * D);
    unsigned short* bufBF = (unsigned short*)alloc(2ull * N * D);
    float*          als   = (float*)alloc(4ull * N * 8);
    float*          ald   = (float*)alloc(4ull * N * 8);
    float*          mseg  = (float*)alloc(4ull * N * 8);
    float*          sseg  = (float*)alloc(4ull * N * 8);
    float*          exbuf = (float*)alloc(4ull * Etot * 8);

    auto nb = [](int n, int b) { return (n + b - 1) / b; };
    const int T = 256;
    const dim3 gemm_grid(D / 64, nb(N / 16, 8));     // (8, 157)

    // ---- one-time casts / repacks ---------------------------------------
    build_edges_kernel<<<nb(Etot, T), T, 0, stream>>>(eidx, srcI, dstI, E, Etot);
    cast_bf16_kernel<<<nb(N * Fin, T), T, 0, stream>>>(x, xb, N * Fin);
    repack_weight_bf16_kernel<<<nb(Fin * D, T), T, 0, stream>>>(W1, w1p, Fin, D);
    repack_weight_bf16_kernel<<<nb(D * D, T), T, 0, stream>>>(fc1_w, fc1p, D, D);
    repack_weight_bf16_kernel<<<nb(D * D, T), T, 0, stream>>>(W2, w2p, D, D);
    repack_weight_bf16_kernel<<<nb(D * D, T), T, 0, stream>>>(fc2_w, fc2p, D, D);

    // ================= GAT layer 1 (H=8, C=64) ===========================
    gemm_bf16_wmma<128><<<gemm_grid, T, 0, stream>>>(xb, w1p, bufA, N, D);
    attn_logits_h8_kernel<<<nb(N * 8, T), T, 0, stream>>>(bufA, a1s, a1d, als, ald, N * 8);
    edge_score_kernel<<<nb(Etot * 8, T), T, 0, stream>>>(srcI, dstI, als, ald, exbuf, Etot, 8);
    fill_f32_kernel<<<nb(N * 8, T), T, 0, stream>>>(mseg, -3.0e38f, N * 8);
    fill_f32_kernel<<<nb(N * 8, T), T, 0, stream>>>(sseg, 0.0f, N * 8);
    seg_max_kernel<<<nb(Etot * 8, T), T, 0, stream>>>(dstI, exbuf, mseg, Etot, 8);
    seg_expsum_kernel<<<nb(Etot * 8, T), T, 0, stream>>>(dstI, exbuf, mseg, sseg, Etot, 8);
    fill_f32_kernel<<<nb(N * D, T), T, 0, stream>>>(bufC, 0.0f, N * D);
    aggregate_kernel<<<Etot, T, 0, stream>>>(srcI, dstI, bufA, exbuf, sseg, bufC, Etot, 8, 64);
    bias_cast_bf16_kernel<<<nb(N * D, T), T, 0, stream>>>(bufC, b1, bufBF, N * D, D);

    // fc1 + ReLU + BN1
    gemm_bf16_wmma<512><<<gemm_grid, T, 0, stream>>>(bufBF, fc1p, bufB, N, D);
    bn_relu_kernel<<<D, T, 0, stream>>>(bufB, fc1_b, g1, be1, bufA, bufBF, N, D);

    // ================= GAT layer 2 (H=1, C=512) ==========================
    gemm_bf16_wmma<512><<<gemm_grid, T, 0, stream>>>(bufBF, w2p, bufB, N, D);
    attn_logits_h1_kernel<<<nb(N, T), T, 0, stream>>>(bufB, a2s, a2d, als, ald, N);
    edge_score_kernel<<<nb(Etot, T), T, 0, stream>>>(srcI, dstI, als, ald, exbuf, Etot, 1);
    fill_f32_kernel<<<nb(N, T), T, 0, stream>>>(mseg, -3.0e38f, N);
    fill_f32_kernel<<<nb(N, T), T, 0, stream>>>(sseg, 0.0f, N);
    seg_max_kernel<<<nb(Etot, T), T, 0, stream>>>(dstI, exbuf, mseg, Etot, 1);
    seg_expsum_kernel<<<nb(Etot, T), T, 0, stream>>>(dstI, exbuf, mseg, sseg, Etot, 1);
    fill_f32_kernel<<<nb(N * D, T), T, 0, stream>>>(bufC, 0.0f, N * D);
    aggregate_kernel<<<Etot, T, 0, stream>>>(srcI, dstI, bufB, exbuf, sseg, bufC, Etot, 1, 512);
    bias_cast_bf16_kernel<<<nb(N * D, T), T, 0, stream>>>(bufC, b2, bufBF, N * D, D);

    // fc2 + ReLU + BN2
    gemm_bf16_wmma<512><<<gemm_grid, T, 0, stream>>>(bufBF, fc2p, bufA, N, D);
    bn_relu_kernel<<<D, T, 0, stream>>>(bufA, fc2_b, g2, be2, bufB, bufBF, N, D);

    // ================= final linear + ReLU ===============================
    final_linear_kernel<<<nb(N * OUT, T), T, 0, stream>>>(bufB, lin_w, lin_b, out, N, OUT);
}
